// RNNDec_89129161327114
// MI455X (gfx1250) — matrix operand order repaired and checked by reading, not compiled, not run on hardware
//
#include <hip/hip_runtime.h>
#include <hip/hip_bf16.h>
#include <math.h>

typedef __attribute__((ext_vector_type(16))) __bf16        v16bf;
typedef __attribute__((ext_vector_type(8)))  float         v8f;
typedef __attribute__((ext_vector_type(8)))  unsigned int  v8u;
typedef __attribute__((ext_vector_type(4)))  int           i32x4;

#define HD   512
#define NB   4096
#define TT   12

__device__ __forceinline__ unsigned short f2bf(float f) {
    unsigned int u = __builtin_bit_cast(unsigned int, f);
    unsigned int r = u + 0x7FFFu + ((u >> 16) & 1u);
    return (unsigned short)(r >> 16);
}

__device__ __forceinline__ float sigmoidf(float x) {
    return 1.0f / (1.0f + __expf(-x));
}

// Async DMA: global -> LDS, 16 bytes per lane, tracked by ASYNCcnt.
__device__ __forceinline__ void async_copy_b128(const void* gptr, void* lptr) {
#if __has_builtin(__builtin_amdgcn_global_load_async_to_lds_b128)
    __builtin_amdgcn_global_load_async_to_lds_b128(
        (__attribute__((address_space(1))) i32x4*)(unsigned long long)gptr,
        (__attribute__((address_space(3))) i32x4*)(unsigned)(unsigned long long)lptr,
        0, 0);
#else
    unsigned ldsoff = (unsigned)(unsigned long long)lptr;   // low 32 bits = LDS offset
    asm volatile("global_load_async_to_lds_b128 %0, %1, off"
                 :: "v"(ldsoff), "v"((unsigned long long)gptr) : "memory");
#endif
}

__device__ __forceinline__ void wait_async0() {
#if __has_builtin(__builtin_amdgcn_s_wait_asynccnt)
    __builtin_amdgcn_s_wait_asynccnt(0);
#else
    asm volatile("s_wait_asynccnt 0x0" ::: "memory");
#endif
}

// ---------------------------------------------------------------------------
// Prep kernels
// ---------------------------------------------------------------------------
__global__ void pack_zx(const float* __restrict__ z, const float* __restrict__ x,
                        unsigned short* __restrict__ zx) {
    int idx = blockIdx.x * blockDim.x + threadIdx.x;        // 4096*256
    if (idx >= NB * 256) return;
    int row = idx >> 8, c = idx & 255;
    float v = (c < 128) ? z[row * 128 + c] : x[row * 128 + (c - 128)];
    zx[idx] = f2bf(v);
}

__global__ void conv_bf16(const float* __restrict__ src, unsigned short* __restrict__ dst, int n) {
    int i = blockIdx.x * blockDim.x + threadIdx.x;
    if (i < n) dst[i] = f2bf(src[i]);
}

// Split W_ih [1536,258]: cols 0..255 -> bf16 [1536,256]; cols 256..257 -> f32 [1536,2]
__global__ void conv_wih(const float* __restrict__ Wih, unsigned short* __restrict__ Wz,
                         float* __restrict__ Wx) {
    int idx = blockIdx.x * blockDim.x + threadIdx.x;        // 1536*258
    if (idx >= 1536 * 258) return;
    int j = idx / 258, c = idx % 258;
    float v = Wih[idx];
    if (c < 256) Wz[j * 256 + c] = f2bf(v);
    else         Wx[j * 2 + (c - 256)] = v;
}

// a0 = x_0 @ W_ia^T + b_ia -> x_buf [4096,2]
__global__ void init_a0(const float* __restrict__ x0, const float* __restrict__ Wia,
                        const float* __restrict__ bia, float* __restrict__ xb) {
    int row = blockIdx.x * blockDim.x + threadIdx.x;
    if (row >= NB) return;
    float v0 = bia[0], v1 = bia[1];
#pragma unroll
    for (int k = 0; k < 4; ++k) {
        float xv = x0[row * 4 + k];
        v0 += xv * Wia[k];
        v1 += xv * Wia[4 + k];
    }
    xb[row * 2 + 0] = v0;
    xb[row * 2 + 1] = v1;
}

// ---------------------------------------------------------------------------
// Generic WMMA GEMM:  C[M,N] = A[M,K](bf16) * B[N,K](bf16)^T + bias[N]
// block = 128 threads (4 waves), tile 32x32, K-chunk 64, double-buffered LDS
// filled by async global->LDS DMA.
// ---------------------------------------------------------------------------
__global__ __launch_bounds__(128) void gemm_bf16_bias(
    const unsigned short* __restrict__ A, const unsigned short* __restrict__ B,
    const float* __restrict__ bias, float* __restrict__ Cf,
    unsigned short* __restrict__ Cbf, int N, int K)
{
    __shared__ __align__(16) unsigned short As[2][32][64];
    __shared__ __align__(16) unsigned short Bs[2][32][64];
    const int tid = threadIdx.x;
    const int wave = tid >> 5, lane = tid & 31;
    const int rowBase = blockIdx.y * 32;
    const int colBase = blockIdx.x * 32;
    const int mw = (wave >> 1) * 16, nw = (wave & 1) * 16;

    const int fm   = mw + (lane & 15);
    const int fn   = nw + (lane & 15);
    const int aoff = (lane >> 4) * 8;   // A: consecutive K per lane-half (uints)
    const int boff = (lane >> 4) * 8;   // B: consecutive K per lane-half (uints)

    v8f acc = {};

    // issue one K-chunk's async copies into LDS buffer `buf`
    auto issue = [&](int buf, int kc) {
#pragma unroll
        for (int i = 0; i < 2; ++i) {   // 32 rows * 8 segs = 256 xfers, 2/thread
            int flat = tid + 128 * i;
            int r = flat >> 3, seg = flat & 7;
            async_copy_b128(A + (size_t)(rowBase + r) * K + kc + seg * 8,
                            &As[buf][r][seg * 8]);
            async_copy_b128(B + (size_t)(colBase + r) * K + kc + seg * 8,
                            &Bs[buf][r][seg * 8]);
        }
    };

    issue(0, 0);
    wait_async0();
    __syncthreads();

    const int nChunks = K >> 6;
    for (int n = 0; n < nChunks; ++n) {
        const int cur = n & 1;
        if (n + 1 < nChunks) issue(cur ^ 1, (n + 1) << 6);

        const unsigned int* au = (const unsigned int*)&As[cur][fm][0];
        const unsigned int* bu = (const unsigned int*)&Bs[cur][fn][0];
#pragma unroll
        for (int s = 0; s < 2; ++s) {   // two 32-K subchunks
            const int base = s * 16;
            v8u aU, bU;
#pragma unroll
            for (int j = 0; j < 4; ++j) {
                aU[j]     = au[base + aoff + j];
                aU[4 + j] = au[base + aoff + 4 + j];
            }
#pragma unroll
            for (int j = 0; j < 8; ++j) bU[j] = bu[base + boff + j];
            v16bf a = __builtin_bit_cast(v16bf, aU);
            v16bf b = __builtin_bit_cast(v16bf, bU);
            acc = __builtin_amdgcn_wmma_f32_16x16x32_bf16(false, a, false, b,
                                                          (short)0, acc, false, false);
        }
        wait_async0();
        __syncthreads();
    }

    const int col   = colBase + nw + (lane & 15);
    const int rbase = rowBase + mw + ((lane >> 4) << 3);
    const float bv  = bias ? bias[col] : 0.0f;
#pragma unroll
    for (int i = 0; i < 8; ++i) {
        const int row = rbase + i;
        float v = acc[i] + bv;
        if (Cf)  Cf[(size_t)row * N + col]  = v;
        if (Cbf) Cbf[(size_t)row * N + col] = f2bf(v);
    }
}

// ---------------------------------------------------------------------------
// GRU step: gh = h @ W_hh^T (3 gate chunks via WMMA) fused with gate math.
// grid = (512/32, 4096/32), block = 128; K-chunk 64, double-buffered async LDS
// ---------------------------------------------------------------------------
__global__ __launch_bounds__(128) void gru_step(
    const unsigned short* __restrict__ Hbf,  // [4096,512] bf16 h_t
    const float* __restrict__ Hf,            // [4096,512] f32 h_t
    const unsigned short* __restrict__ Whh,  // [1536,512] bf16
    const float* __restrict__ bhh,           // [1536]
    const float* __restrict__ gi_zx,         // [4096,1536] (b_ih folded in)
    const float* __restrict__ WihX,          // [1536,2]
    const float* __restrict__ xt,            // [4096,2]
    float* __restrict__ Hf_new,
    unsigned short* __restrict__ Hbf_new)
{
    __shared__ __align__(16) unsigned short As[2][32][64];
    __shared__ __align__(16) unsigned short Bs[2][3][32][64];
    const int tid = threadIdx.x;
    const int wave = tid >> 5, lane = tid & 31;
    const int rowBase = blockIdx.y * 32;
    const int colBase = blockIdx.x * 32;
    const int mw = (wave >> 1) * 16, nw = (wave & 1) * 16;

    const int fm   = mw + (lane & 15);
    const int fn   = nw + (lane & 15);
    const int aoff = (lane >> 4) * 8;
    const int boff = (lane >> 4) * 8;

    v8f accr = {}, accz = {}, accn = {};

    auto issue = [&](int buf, int kc) {
#pragma unroll
        for (int i = 0; i < 2; ++i) {   // A: 256 xfers
            int flat = tid + 128 * i;
            int r = flat >> 3, seg = flat & 7;
            async_copy_b128(Hbf + (size_t)(rowBase + r) * HD + kc + seg * 8,
                            &As[buf][r][seg * 8]);
        }
#pragma unroll
        for (int i = 0; i < 6; ++i) {   // B: 3 gates * 32 rows * 8 segs = 768 xfers
            int flat = tid + 128 * i;
            int brow = flat >> 3, seg = flat & 7;
            int g = brow >> 5, nn = brow & 31;
            async_copy_b128(Whh + (size_t)(colBase + g * HD + nn) * HD + kc + seg * 8,
                            &Bs[buf][g][nn][seg * 8]);
        }
    };

    issue(0, 0);
    wait_async0();
    __syncthreads();

    const int nChunks = HD >> 6;        // 8
    for (int n = 0; n < nChunks; ++n) {
        const int cur = n & 1;
        if (n + 1 < nChunks) issue(cur ^ 1, (n + 1) << 6);

        const unsigned int* au = (const unsigned int*)&As[cur][fm][0];
#pragma unroll
        for (int s = 0; s < 2; ++s) {
            const int base = s * 16;
            v8u aU;
#pragma unroll
            for (int j = 0; j < 4; ++j) {
                aU[j]     = au[base + aoff + j];
                aU[4 + j] = au[base + aoff + 4 + j];
            }
            v16bf a = __builtin_bit_cast(v16bf, aU);
#pragma unroll
            for (int g = 0; g < 3; ++g) {
                const unsigned int* bu = (const unsigned int*)&Bs[cur][g][fn][0];
                v8u bU;
#pragma unroll
                for (int j = 0; j < 8; ++j) bU[j] = bu[base + boff + j];
                v16bf b = __builtin_bit_cast(v16bf, bU);
                if (g == 0)
                    accr = __builtin_amdgcn_wmma_f32_16x16x32_bf16(false, a, false, b,
                                                                   (short)0, accr, false, false);
                else if (g == 1)
                    accz = __builtin_amdgcn_wmma_f32_16x16x32_bf16(false, a, false, b,
                                                                   (short)0, accz, false, false);
                else
                    accn = __builtin_amdgcn_wmma_f32_16x16x32_bf16(false, a, false, b,
                                                                   (short)0, accn, false, false);
            }
        }
        wait_async0();
        __syncthreads();
    }

    const int col   = colBase + nw + (lane & 15);
    const int rbase = rowBase + mw + ((lane >> 4) << 3);
    const float bhr = bhh[col], bhz = bhh[col + 512], bhn = bhh[col + 1024];
    const float wr0 = WihX[col * 2],            wr1 = WihX[col * 2 + 1];
    const float wz0 = WihX[(col + 512) * 2],    wz1 = WihX[(col + 512) * 2 + 1];
    const float wn0 = WihX[(col + 1024) * 2],   wn1 = WihX[(col + 1024) * 2 + 1];

#pragma unroll
    for (int i = 0; i < 8; ++i) {
        const int row = rbase + i;
        const size_t gbase = (size_t)row * 1536 + col;
        const float x0 = xt[row * 2], x1 = xt[row * 2 + 1];
        const float gir = gi_zx[gbase]        + x0 * wr0 + x1 * wr1;
        const float giz = gi_zx[gbase + 512]  + x0 * wz0 + x1 * wz1;
        const float gin = gi_zx[gbase + 1024] + x0 * wn0 + x1 * wn1;
        const float r = sigmoidf(gir + accr[i] + bhr);
        const float u = sigmoidf(giz + accz[i] + bhz);
        const float nn = tanhf(gin + r * (accn[i] + bhn));
        const float ho = Hf[(size_t)row * HD + col];
        const float h  = (1.0f - u) * nn + u * ho;
        Hf_new[(size_t)row * HD + col]  = h;
        Hbf_new[(size_t)row * HD + col] = f2bf(h);
    }
}

// ---------------------------------------------------------------------------
// out = h_new @ W_out^T + b_out ; also feeds x_{t+1}. 8 rows/block, wave/row.
// ---------------------------------------------------------------------------
__global__ __launch_bounds__(256) void out_proj(
    const float* __restrict__ H, const float* __restrict__ Wout,
    const float* __restrict__ bout, float* __restrict__ out,
    float* __restrict__ x_next, int t)
{
    const int wave = threadIdx.x >> 5, lane = threadIdx.x & 31;
    const int row = blockIdx.x * 8 + wave;
    float s0 = 0.0f, s1 = 0.0f;
    for (int c = lane; c < HD; c += 32) {
        float h = H[(size_t)row * HD + c];
        s0 += h * Wout[c];
        s1 += h * Wout[HD + c];
    }
#pragma unroll
    for (int off = 16; off > 0; off >>= 1) {
        s0 += __shfl_xor(s0, off, 32);
        s1 += __shfl_xor(s1, off, 32);
    }
    if (lane == 0) {
        float o0 = s0 + bout[0], o1 = s1 + bout[1];
        out[((size_t)row * TT + t) * 2 + 0] = o0;
        out[((size_t)row * TT + t) * 2 + 1] = o1;
        x_next[row * 2 + 0] = o0;
        x_next[row * 2 + 1] = o1;
    }
}

// ---------------------------------------------------------------------------
extern "C" void kernel_launch(void* const* d_in, const int* in_sizes, int n_in,
                              void* d_out, int out_size, void* d_ws, size_t ws_size,
                              hipStream_t stream) {
    const float* z    = (const float*)d_in[0];
    const float* x    = (const float*)d_in[1];
    const float* x0   = (const float*)d_in[2];
    const float* Wia  = (const float*)d_in[3];
    const float* bia  = (const float*)d_in[4];
    const float* Wh0  = (const float*)d_in[5];
    const float* bh0  = (const float*)d_in[6];
    const float* Wih  = (const float*)d_in[7];
    const float* bih  = (const float*)d_in[8];
    const float* Whh  = (const float*)d_in[9];
    const float* bhh  = (const float*)d_in[10];
    const float* Wout = (const float*)d_in[11];
    const float* bout = (const float*)d_in[12];
    float* out = (float*)d_out;

    char* p = (char*)d_ws;
    unsigned short* zx_bf  = (unsigned short*)p; p += (size_t)NB * 256 * 2;
    unsigned short* h_bf0  = (unsigned short*)p; p += (size_t)NB * HD * 2;
    unsigned short* h_bf1  = (unsigned short*)p; p += (size_t)NB * HD * 2;
    float*          h_f0   = (float*)p;          p += (size_t)NB * HD * 4;
    float*          h_f1   = (float*)p;          p += (size_t)NB * HD * 4;
    unsigned short* Wh0_bf = (unsigned short*)p; p += (size_t)HD * 256 * 2;
    unsigned short* Wih_bf = (unsigned short*)p; p += (size_t)1536 * 256 * 2;
    float*          WihX   = (float*)p;          p += (size_t)1536 * 2 * 4;
    unsigned short* Whh_bf = (unsigned short*)p; p += (size_t)1536 * HD * 2;
    float*          x_buf  = (float*)p;          p += (size_t)NB * 2 * 4;
    float*          gi_zx  = (float*)p;          p += (size_t)NB * 1536 * 4;

    unsigned short* h_bf[2] = { h_bf0, h_bf1 };
    float*          h_f[2]  = { h_f0,  h_f1  };

    // --- prep ---
    pack_zx <<<(NB * 256 + 255) / 256, 256, 0, stream>>>(z, x, zx_bf);
    conv_bf16<<<(HD * 256 + 255) / 256, 256, 0, stream>>>(Wh0, Wh0_bf, HD * 256);
    conv_wih <<<(1536 * 258 + 255) / 256, 256, 0, stream>>>(Wih, Wih_bf, WihX);
    conv_bf16<<<(1536 * HD + 255) / 256, 256, 0, stream>>>(Whh, Whh_bf, 1536 * HD);
    init_a0  <<<(NB + 255) / 256, 256, 0, stream>>>(x0, Wia, bia, x_buf);

    // h0 = zx @ W_h0^T + b_h0   (M=4096, N=512, K=256)
    gemm_bf16_bias<<<dim3(HD / 32, NB / 32), 128, 0, stream>>>(
        zx_bf, Wh0_bf, bh0, h_f[0], h_bf[0], HD, 256);

    // gi_zx = zx @ W_ih[:, :256]^T + b_ih   (M=4096, N=1536, K=256)
    gemm_bf16_bias<<<dim3(1536 / 32, NB / 32), 128, 0, stream>>>(
        zx_bf, Wih_bf, bih, gi_zx, (unsigned short*)nullptr, 1536, 256);

    // --- 12 recurrent steps ---
    for (int t = 0; t < TT; ++t) {
        int cur = t & 1, nxt = 1 - cur;
        gru_step<<<dim3(HD / 32, NB / 32), 128, 0, stream>>>(
            h_bf[cur], h_f[cur], Whh_bf, bhh, gi_zx, WihX, x_buf,
            h_f[nxt], h_bf[nxt]);
        out_proj<<<NB / 8, 256, 0, stream>>>(h_f[nxt], Wout, bout, out, x_buf, t);
    }
}
